// TemporalGNN_Base_21680994910533
// MI455X (gfx1250) — compile-verified, matrix operand order checked by reading, not run on hardware
//
#include <hip/hip_runtime.h>

// ---------------------------------------------------------------------------
// TemporalGNN for MI455X (gfx1250, wave32, WMMA).
// GRU recurrence + 2x SAGEConv. All GEMMs via v_wmma_f32_16x16x32_f16.
// Weight staging via global_load_async_to_lds_b128 (ASYNCcnt path).
// ---------------------------------------------------------------------------

typedef __attribute__((ext_vector_type(16))) _Float16 v16h;
typedef __attribute__((ext_vector_type(8)))  float    v8f;

constexpr int kT   = 24;
constexpr int kF   = 16;    // F_IN
constexpr int kH   = 128;   // hidden
constexpr int kG3  = 384;   // 3*H
constexpr int kOUT = 64;
constexpr int NTILE = 32;   // nodes per GRU block

// ---- gfx1250 async copy: global -> LDS, 16B per lane, tracked on ASYNCcnt.
// LDS operand is the wave-relative LDS byte offset == low 32 bits of the
// flat pointer.
__device__ __forceinline__ unsigned lds_off_of(const void* p) {
  return (unsigned)(uintptr_t)p;
}
__device__ __forceinline__ void async_g2l_b128(unsigned lds_off,
                                               const void* gaddr) {
  asm volatile("global_load_async_to_lds_b128 %0, %1, off"
               :: "v"(lds_off), "v"(gaddr) : "memory");
}
__device__ __forceinline__ void wait_async0() {
  asm volatile("s_wait_asynccnt 0" ::: "memory");
}

// ---- fast transcendentals (co-execute with WMMA on the TRANS pipe) ----
__device__ __forceinline__ float fast_sigmoid(float x) {
  return __builtin_amdgcn_rcpf(1.f + __expf(-x));
}
__device__ __forceinline__ float fast_tanh(float x) {
#if __has_builtin(__builtin_amdgcn_tanhf)
  return __builtin_amdgcn_tanhf(x);           // gfx1250 v_tanh_f32
#else
  float e = __expf(2.f * x);
  return 1.f - 2.f * __builtin_amdgcn_rcpf(e + 1.f);
#endif
}

// ---- WMMA fragment helpers (16-bit A/B layout: lane&15 = row/col,
// lane&16 selects K half-group; per lane two contiguous 8-half runs) ----
__device__ __forceinline__ v16h ld_frag(const _Float16* p0, int ldk,
                                        int row0, int k0, int lane) {
  union { v16h v; _Float16 h[16]; } f;
  const _Float16* p = p0 + (size_t)(row0 + (lane & 15)) * ldk + k0
                         + ((lane & 16) ? 8 : 0);
#pragma unroll
  for (int j = 0; j < 4; ++j) { f.h[2*j] = p[2*j]; f.h[2*j+1] = p[2*j+1]; }
#pragma unroll
  for (int j = 4; j < 8; ++j) {
    int kb = 16 + 2*(j - 4);
    f.h[2*j] = p[kb]; f.h[2*j+1] = p[kb+1];
  }
  return f.v;
}

// K=16 matrix padded to K=32 (upper 16 K-values are zero)
__device__ __forceinline__ v16h ld_frag_pad16(const _Float16* p0, int ldk,
                                              int row0, int lane) {
  union { v16h v; _Float16 h[16]; } f;
  const _Float16* p = p0 + (size_t)(row0 + (lane & 15)) * ldk
                         + ((lane & 16) ? 8 : 0);
#pragma unroll
  for (int j = 0; j < 4; ++j) { f.h[2*j] = p[2*j]; f.h[2*j+1] = p[2*j+1]; }
#pragma unroll
  for (int j = 4; j < 8; ++j) { f.h[2*j] = (_Float16)0.f; f.h[2*j+1] = (_Float16)0.f; }
  return f.v;
}

// 16x16 f32 C/D tile -> LDS (VGPR r: lanes 0-15 M=r, lanes 16-31 M=8+r)
__device__ __forceinline__ void st_c(float* p0, int ldn, int m0, int n0,
                                     int lane, v8f c) {
  float* p = p0 + (size_t)(m0 + ((lane & 16) ? 8 : 0)) * ldn + n0 + (lane & 15);
#pragma unroll
  for (int r = 0; r < 8; ++r) p[(size_t)r * ldn] = c[r];
}

// ---------------------------------------------------------------------------
// f32 -> f16 weight conversion
// ---------------------------------------------------------------------------
__global__ void f32_to_f16_kernel(const float* __restrict__ in,
                                  _Float16* __restrict__ out, int n) {
  int i = blockIdx.x * blockDim.x + threadIdx.x;
  if (i < n) out[i] = (_Float16)in[i];
}

// ---------------------------------------------------------------------------
// GRU: block = 32 nodes, 256 threads (8 waves). All state in LDS/registers.
// Per step: gates[32x384] = h@Whh.T + x_t@Wih.T as 16x16 WMMA tiles.
// Combined accumulator gi+gh for all gates; separate gi_n for the n-gate.
// ---------------------------------------------------------------------------
__global__ __launch_bounds__(256) void gru_kernel(
    const float* __restrict__ x,          // [N, T, F]
    const _Float16* __restrict__ Whh16,   // [384,128]
    const _Float16* __restrict__ Wih16,   // [384,16]
    const float* __restrict__ bih, const float* __restrict__ bhh,
    float* __restrict__ hout, _Float16* __restrict__ hout16, int N) {
  __shared__ __align__(16) _Float16 sWhh[kG3][kH];     // 98304 B
  __shared__ __align__(16) _Float16 sWih[kG3][kF];     // 12288 B
  __shared__ __align__(16) _Float16 sH16[NTILE][kH];   //  8192 B
  __shared__ __align__(16) _Float16 sX[NTILE][kT][kF]; // 24576 B
  __shared__ float    sSum[NTILE][kG3];     // 49152 B  gi+gh (no bias)
  __shared__ float    sGIn[NTILE][kH];      // 16384 B  gi for n-gate only
  __shared__ float    sBsum[kG3];           // bih+bhh
  __shared__ float    sBihn[kH];            // bih n-part

  const int tid = threadIdx.x;
  const int node0 = blockIdx.x * NTILE;

  // async weight staging: bytes go straight to LDS, no VGPR round-trip
  {
    const unsigned whh0 = lds_off_of(&sWhh[0][0]);
    for (int i = tid; i < (kG3 * kH) / 8; i += 256)      // 16B chunks
      async_g2l_b128(whh0 + i * 16, Whh16 + i * 8);
    const unsigned wih0 = lds_off_of(&sWih[0][0]);
    for (int i = tid; i < (kG3 * kF) / 8; i += 256)
      async_g2l_b128(wih0 + i * 16, Wih16 + i * 8);
  }
  for (int i = tid; i < kG3; i += 256) sBsum[i] = bih[i] + bhh[i];
  for (int i = tid; i < kH; i += 256) sBihn[i] = bih[2 * kH + i];
  for (int i = tid; i < NTILE * kT * kF; i += 256) {
    int n = i / (kT * kF);
    int rem = i - n * (kT * kF);
    int g = node0 + n;
    float v = (g < N) ? x[(size_t)g * (kT * kF) + rem] : 0.f;
    (&sX[0][0][0])[i] = (_Float16)v;
  }
  for (int i = tid; i < NTILE * kH; i += 256) (&sH16[0][0])[i] = (_Float16)0.f;

  float hreg[(NTILE * kH) / 256];           // f32 master copy of h
#pragma unroll
  for (int k = 0; k < (NTILE * kH) / 256; ++k) hreg[k] = 0.f;

  wait_async0();
  __syncthreads();

  const int wave = tid >> 5, lane = tid & 31;

  for (int t = 0; t < kT; ++t) {
    const _Float16* xt = &sX[0][t][0];      // row stride kT*kF between nodes
    // ---- 48 output tiles (2 node-subtiles x 24 gate-chunks) over 8 waves
    for (int tile = wave; tile < 2 * (kG3 / 16); tile += 8) {
      const int m0 = (tile & 1) * 16;
      const int nt = tile >> 1;
      const int n0 = nt * 16;
      v8f acc = {};
#pragma unroll
      for (int kk = 0; kk < kH / 32; ++kk) {
        v16h a = ld_frag(&sH16[0][0], kH, m0, kk * 32, lane);
        v16h b = ld_frag(&sWhh[0][0], kH, n0, kk * 32, lane);
        acc = __builtin_amdgcn_wmma_f32_16x16x32_f16(
            false, a, false, b, (short)0, acc, false, false);
      }
      // fold the input projection (K=16 padded to 32) into the same acc
      {
        v16h a = ld_frag_pad16(xt, kT * kF, m0, lane);
        v16h b = ld_frag_pad16(&sWih[0][0], kF, n0, lane);
        acc = __builtin_amdgcn_wmma_f32_16x16x32_f16(
            false, a, false, b, (short)0, acc, false, false);
      }
      st_c(&sSum[0][0], kG3, m0, n0, lane, acc);
      if (nt >= 16) {                       // n-gate chunks: keep gi_n alone
        v8f gin = {};
        v16h a = ld_frag_pad16(xt, kT * kF, m0, lane);
        v16h b = ld_frag_pad16(&sWih[0][0], kF, n0, lane);
        gin = __builtin_amdgcn_wmma_f32_16x16x32_f16(
            false, a, false, b, (short)0, gin, false, false);
        st_c(&sGIn[0][0], kH, m0, n0 - 2 * kH, lane, gin);
      }
    }
    __syncthreads();
    // ---- elementwise gate update: each thread owns fixed (m,j) slots
#pragma unroll
    for (int k = 0; k < (NTILE * kH) / 256; ++k) {
      int i = tid + k * 256;
      int m = i >> 7, j = i & (kH - 1);
      float r = fast_sigmoid(sSum[m][j] + sBsum[j]);
      float z = fast_sigmoid(sSum[m][kH + j] + sBsum[kH + j]);
      float gin = sGIn[m][j];
      float ghn = sSum[m][2 * kH + j] - gin;
      float bihn = sBihn[j];
      float bhhn = sBsum[2 * kH + j] - bihn;
      float nn = fast_tanh(gin + bihn + r * (ghn + bhhn));
      float hn = (1.f - z) * nn + z * hreg[k];
      hreg[k] = hn;
      sH16[m][j] = (_Float16)hn;
    }
    __syncthreads();
  }

#pragma unroll
  for (int k = 0; k < (NTILE * kH) / 256; ++k) {
    int i = tid + k * 256;
    int m = i >> 7, j = i & (kH - 1);
    int g = node0 + m;
    if (g < N) {
      hout[(size_t)g * kH + j] = hreg[k];
      hout16[(size_t)g * kH + j] = (_Float16)hreg[k];
    }
  }
}

// ---------------------------------------------------------------------------
// Edge degree + scatter-add (L2-resident: h/agg arrays fit in 192MB L2)
// ---------------------------------------------------------------------------
__global__ void count_kernel(const long long* __restrict__ eidx,
                             float* __restrict__ cnt, int E) {
  int e = blockIdx.x * blockDim.x + threadIdx.x;
  if (e < E) atomicAdd(&cnt[(int)eidx[(size_t)E + e]], 1.0f);
}

__global__ void scatter_kernel(const long long* __restrict__ eidx,
                               const float* __restrict__ feat,
                               float* __restrict__ agg, int E) {
  long long idx = (long long)blockIdx.x * blockDim.x + threadIdx.x;
  const int groups = kH / 4;
  if (idx >= (long long)E * groups) return;
  int e = (int)(idx / groups);
  int c = (int)(idx - (long long)e * groups);
  int src = (int)eidx[e];
  int dst = (int)eidx[(size_t)E + e];
  float4 v = *(const float4*)(feat + (size_t)src * kH + c * 4);
  float* o = agg + (size_t)dst * kH + c * 4;
  atomicAdd(o + 0, v.x); atomicAdd(o + 1, v.y);
  atomicAdd(o + 2, v.z); atomicAdd(o + 3, v.w);
}

// ---------------------------------------------------------------------------
// SAGE linear: out = mean @ Wl.T + b + x @ Wr.T  (single WMMA accumulator)
// optional fused residual + ReLU (layer 1)
// ---------------------------------------------------------------------------
template <int NOUT, bool RELU_RES>
__global__ __launch_bounds__(256) void sage_linear_kernel(
    const float* __restrict__ agg, const float* __restrict__ cnt,
    const _Float16* __restrict__ x16, const float* __restrict__ xf,
    const _Float16* __restrict__ Wl16, const _Float16* __restrict__ Wr16,
    const float* __restrict__ bl,
    float* __restrict__ out, _Float16* __restrict__ out16, int N) {
  __shared__ __align__(16) _Float16 sWl[NOUT][kH];
  __shared__ __align__(16) _Float16 sWr[NOUT][kH];
  __shared__ __align__(16) _Float16 sM[16][kH];
  __shared__ __align__(16) _Float16 sXl[16][kH];
  __shared__ float    sOut[16][NOUT];

  const int tid = threadIdx.x;
  const int node0 = blockIdx.x * 16;

  {
    const unsigned wl0 = lds_off_of(&sWl[0][0]);
    const unsigned wr0 = lds_off_of(&sWr[0][0]);
    for (int i = tid; i < (NOUT * kH) / 8; i += blockDim.x) {
      async_g2l_b128(wl0 + i * 16, Wl16 + i * 8);
      async_g2l_b128(wr0 + i * 16, Wr16 + i * 8);
    }
  }
  for (int i = tid; i < 16 * kH; i += blockDim.x) {
    int m = i >> 7, j = i & (kH - 1);
    int g = node0 + m;
    float mv = 0.f, xv = 0.f;
    if (g < N) {
      float c = cnt[g]; c = (c > 1.f) ? c : 1.f;
      mv = agg[(size_t)g * kH + j] * __builtin_amdgcn_rcpf(c);
      xv = (float)x16[(size_t)g * kH + j];
    }
    sM[m][j] = (_Float16)mv;
    sXl[m][j] = (_Float16)xv;
  }
  wait_async0();
  __syncthreads();

  const int wave = tid >> 5, lane = tid & 31;
  const int nwaves = blockDim.x >> 5;
  for (int nt = wave; nt < NOUT / 16; nt += nwaves) {
    const int n0 = nt * 16;
    v8f acc = {};
#pragma unroll
    for (int kk = 0; kk < kH / 32; ++kk) {
      v16h a = ld_frag(&sM[0][0], kH, 0, kk * 32, lane);
      v16h b = ld_frag(&sWl[0][0], kH, n0, kk * 32, lane);
      acc = __builtin_amdgcn_wmma_f32_16x16x32_f16(
          false, a, false, b, (short)0, acc, false, false);
    }
#pragma unroll
    for (int kk = 0; kk < kH / 32; ++kk) {
      v16h a = ld_frag(&sXl[0][0], kH, 0, kk * 32, lane);
      v16h b = ld_frag(&sWr[0][0], kH, n0, kk * 32, lane);
      acc = __builtin_amdgcn_wmma_f32_16x16x32_f16(
          false, a, false, b, (short)0, acc, false, false);
    }
    st_c(&sOut[0][0], NOUT, 0, n0, lane, acc);
  }
  __syncthreads();

  for (int i = tid; i < 16 * NOUT; i += blockDim.x) {
    int m = i / NOUT, j = i - m * NOUT;
    int g = node0 + m;
    if (g < N) {
      float v = sOut[m][j] + bl[j];
      if (RELU_RES) {
        v += xf[(size_t)g * kH + j];       // residual (NOUT == kH)
        v = (v > 0.f) ? v : 0.f;
      }
      out[(size_t)g * NOUT + j] = v;
      if (out16) out16[(size_t)g * NOUT + j] = (_Float16)v;
    }
  }
}

// ---------------------------------------------------------------------------
extern "C" void kernel_launch(void* const* d_in, const int* in_sizes, int n_in,
                              void* d_out, int out_size, void* d_ws,
                              size_t ws_size, hipStream_t stream) {
  (void)n_in; (void)out_size; (void)ws_size;
  const float*     x    = (const float*)d_in[0];
  const long long* eidx = (const long long*)d_in[1];   // int64 edge_index
  const float* Wih = (const float*)d_in[2];
  const float* Whh = (const float*)d_in[3];
  const float* bih = (const float*)d_in[4];
  const float* bhh = (const float*)d_in[5];
  const float* W1l = (const float*)d_in[6];
  const float* b1l = (const float*)d_in[7];
  const float* W1r = (const float*)d_in[8];
  const float* W2l = (const float*)d_in[9];
  const float* b2l = (const float*)d_in[10];
  const float* W2r = (const float*)d_in[11];

  const int N = in_sizes[0] / (kT * kF);
  const int E = in_sizes[1] / 2;

  // ---- workspace carve-out (~210 MB) ----
  char* base = (char*)d_ws;
  size_t off = 0;
  auto alloc = [&](size_t bytes) -> void* {
    void* p = base + off;
    off += (bytes + 255) & ~(size_t)255;
    return p;
  };
  float*     h_f32 = (float*)alloc((size_t)N * kH * 4);
  _Float16*  h_f16 = (_Float16*)alloc((size_t)N * kH * 2);
  float*     g_f32 = (float*)alloc((size_t)N * kH * 4);
  _Float16*  g_f16 = (_Float16*)alloc((size_t)N * kH * 2);
  float*     agg   = (float*)alloc((size_t)N * kH * 4);
  float*     cnt   = (float*)alloc((size_t)N * 4);
  _Float16* Whh16 = (_Float16*)alloc((size_t)kG3 * kH * 2);
  _Float16* Wih16 = (_Float16*)alloc((size_t)kG3 * kF * 2);
  _Float16* W1l16 = (_Float16*)alloc((size_t)kH * kH * 2);
  _Float16* W1r16 = (_Float16*)alloc((size_t)kH * kH * 2);
  _Float16* W2l16 = (_Float16*)alloc((size_t)kOUT * kH * 2);
  _Float16* W2r16 = (_Float16*)alloc((size_t)kOUT * kH * 2);

  auto cvt = [&](const float* src, _Float16* dst, int n) {
    f32_to_f16_kernel<<<(n + 255) / 256, 256, 0, stream>>>(src, dst, n);
  };
  cvt(Whh, Whh16, kG3 * kH);
  cvt(Wih, Wih16, kG3 * kF);
  cvt(W1l, W1l16, kH * kH);
  cvt(W1r, W1r16, kH * kH);
  cvt(W2l, W2l16, kOUT * kH);
  cvt(W2r, W2r16, kOUT * kH);

  // ---- GRU over T=24 steps, fully LDS-resident per 32-node tile ----
  gru_kernel<<<(N + NTILE - 1) / NTILE, 256, 0, stream>>>(
      x, Whh16, Wih16, bih, bhh, h_f32, h_f16, N);

  // ---- SAGE layer 1 ----
  hipMemsetAsync(agg, 0, (size_t)N * kH * 4, stream);
  hipMemsetAsync(cnt, 0, (size_t)N * 4, stream);
  count_kernel<<<(E + 255) / 256, 256, 0, stream>>>(eidx, cnt, E);
  {
    long long total = (long long)E * (kH / 4);
    scatter_kernel<<<(int)((total + 255) / 256), 256, 0, stream>>>(
        eidx, h_f32, agg, E);
  }
  sage_linear_kernel<kH, true><<<(N + 15) / 16, 256, 0, stream>>>(
      agg, cnt, h_f16, h_f32, W1l16, W1r16, b1l, g_f32, g_f16, N);

  // ---- SAGE layer 2 (same edges -> reuse cnt) ----
  hipMemsetAsync(agg, 0, (size_t)N * kH * 4, stream);
  {
    long long total = (long long)E * (kH / 4);
    scatter_kernel<<<(int)((total + 255) / 256), 256, 0, stream>>>(
        eidx, g_f32, agg, E);
  }
  sage_linear_kernel<kOUT, false><<<(N + 15) / 16, 128, 0, stream>>>(
      agg, cnt, g_f16, nullptr, W2l16, W2r16, b2l, (float*)d_out, nullptr, N);
}
